// LipophilicityGNN_65532611002535
// MI455X (gfx1250) — matrix operand order repaired and checked by reading, not compiled, not run on hardware
//
#include <hip/hip_runtime.h>

#define N_NODES  100000
#define N_EDGES  1600000
#define IN_DIM   64
#define HID      128
#define N_GRAPHS 2048

typedef __attribute__((ext_vector_type(16))) __bf16 v16bf;
typedef __attribute__((ext_vector_type(8)))  float  v8f;

union Frag16 { uint4 q[2]; v16bf v; };

__device__ __forceinline__ unsigned short f2bf(float f) {
    unsigned u = __builtin_bit_cast(unsigned, f);
    unsigned r = u + 0x7FFFu + ((u >> 16) & 1u);   // round-to-nearest-even
    return (unsigned short)(r >> 16);
}

// ---------------- utility kernels ----------------
__global__ void k_fill(float* p, float v, long long n) {
    long long i = (long long)blockIdx.x * blockDim.x + threadIdx.x;
    if (i < n) p[i] = v;
}

__global__ void k_count_deg(const int* dst, float* deg, int E) {
    int e = blockIdx.x * blockDim.x + threadIdx.x;
    if (e < E) atomicAdd(&deg[dst[e]], 1.0f);
}

__global__ void k_rsqrt_inplace(float* deg, int n) {
    int i = blockIdx.x * blockDim.x + threadIdx.x;
    if (i < n) deg[i] = rsqrtf(deg[i]);   // deg >= 1 due to self loop
}

__global__ void k_cast_bf16(const float* x, unsigned short* xb, long long n) {
    long long i = (long long)blockIdx.x * blockDim.x + threadIdx.x;
    if (i < n) xb[i] = f2bf(x[i]);
}

// Pack row-major weight W[K,128] into WMMA B-fragment layout.
// Fragment (ct, s): 32 lanes x 8 dwords; lane<16 -> col=ct*16+lane, K=s*32+2i{,+1}
//                   lane>=16 -> col=ct*16+lane-16, K=s*32+16+2i{,+1}
__global__ void k_pack_w(const float* W, unsigned* Bp, int Ksteps) {
    int idx = blockIdx.x * blockDim.x + threadIdx.x;
    int total = 8 * Ksteps * 32 * 8;
    if (idx >= total) return;
    int i    = idx & 7;
    int lane = (idx >> 3) & 31;
    int rest = idx >> 8;                 // rest = ct*Ksteps + s
    int s    = rest % Ksteps;
    int ct   = rest / Ksteps;
    int col  = ct * 16 + (lane & 15);
    int k    = s * 32 + ((lane >= 16) ? 16 : 0) + 2 * i;
    unsigned lo = f2bf(W[(long long)k * HID + col]);
    unsigned hi = f2bf(W[(long long)(k + 1) * HID + col]);
    Bp[idx] = lo | (hi << 16);
}

// ---------------- WMMA GEMM: C[M,128] = A[M,K](bf16) @ W(packed bf16) ----------------
// One wave computes a 16-row x 128-col tile; 8 waves per block.
__global__ void k_gemm_bf16(const unsigned short* __restrict__ A,
                            const unsigned* __restrict__ Bp,
                            float* __restrict__ C,
                            int Mtiles, int Ksteps) {
    int wave = threadIdx.x >> 5;
    int lane = threadIdx.x & 31;
    int mtile = blockIdx.x * 8 + wave;
    if (mtile >= Mtiles) return;                 // wave-uniform; EXEC stays all-1s

    int half = lane >> 4;                        // K-half select for A fragment
    int m    = mtile * 16 + (lane & 15);
    int K    = Ksteps * 32;
    const unsigned short* arow = A + (long long)m * K;

    v8f acc[8] = {};

    for (int s = 0; s < Ksteps; ++s) {
        int kb = s * 32;
        Frag16 a;
        a.q[0] = *reinterpret_cast<const uint4*>(arow + kb + half * 16);       // K: kb+16h .. +7
        a.q[1] = *reinterpret_cast<const uint4*>(arow + kb + half * 16 + 8);   // K: kb+16h+8 .. +7
#pragma unroll
        for (int ct = 0; ct < 8; ++ct) {
            const uint4* bp = reinterpret_cast<const uint4*>(
                Bp + (((long long)(ct * Ksteps + s)) * 32 + lane) * 8);
            Frag16 b;
            b.q[0] = bp[0];
            b.q[1] = bp[1];
            acc[ct] = __builtin_amdgcn_wmma_f32_16x16x32_bf16(
                false, a.v, false, b.v, (short)0, acc[ct], false, false);
        }
    }

    int n0 = lane & 15;
    int mrow = mtile * 16 + 8 * half;            // VGPR r -> row mrow + r
#pragma unroll
    for (int ct = 0; ct < 8; ++ct)
#pragma unroll
        for (int r = 0; r < 8; ++r)
            C[(long long)(mrow + r) * HID + ct * 16 + n0] = acc[ct][r];
}

// ---------------- GCN propagation ----------------
// Self-loop term initializes H: H[i] = T[i] * dinv[i]^2
__global__ void k_selfloop(const float* __restrict__ T, const float* __restrict__ dinv,
                           float* __restrict__ H, long long total) {
    long long idx = (long long)blockIdx.x * blockDim.x + threadIdx.x;
    if (idx >= total) return;
    long long i = idx >> 7;
    float di = dinv[i];
    H[idx] = T[idx] * di * di;
}

// Edge scatter: H[dst] += T[src] * dinv[src]*dinv[dst]  (f32 atomics, L2-resident)
__global__ void k_edge_agg(const int* __restrict__ src, const int* __restrict__ dst,
                           const float* __restrict__ dinv,
                           const float* __restrict__ T, float* __restrict__ H, int E) {
    int f   = threadIdx.x & 127;
    int sub = threadIdx.x >> 7;
    int e0  = blockIdx.x * 8;
    for (int j = sub; j < 8; j += 2) {
        int e = e0 + j;
        if (e >= E) continue;
        int s = src[e], d = dst[e];
        float nrm = dinv[s] * dinv[d];
        float v = T[(long long)s * HID + f] * nrm;
        atomicAdd(&H[(long long)d * HID + f], v);
    }
}

// H = relu(H + bias[f]); optionally emit bf16 copy for next GEMM
__global__ void k_bias_relu_cast(float* __restrict__ H, const float* __restrict__ bias,
                                 unsigned short* __restrict__ Hb, long long total) {
    long long idx = (long long)blockIdx.x * blockDim.x + threadIdx.x;
    if (idx >= total) return;
    int f = (int)(idx & 127);
    float h = H[idx] + bias[f];
    h = fmaxf(h, 0.0f);
    H[idx] = h;
    if (Hb) Hb[idx] = f2bf(h);
}

// ---------------- pooling + FC ----------------
__global__ void k_pool(const int* __restrict__ batch, const float* __restrict__ H,
                       float* __restrict__ sums, float* __restrict__ cnt, int Nn) {
    int f   = threadIdx.x & 127;
    int sub = threadIdx.x >> 7;
    int n0  = blockIdx.x * 8;
    for (int j = sub; j < 8; j += 2) {
        int node = n0 + j;
        if (node >= Nn) continue;
        int b = batch[node];
        atomicAdd(&sums[(long long)b * HID + f], H[(long long)node * HID + f]);
        if (f == 0) atomicAdd(&cnt[b], 1.0f);
    }
}

__global__ void k_mean_cast(const float* __restrict__ sums, const float* __restrict__ cnt,
                            unsigned short* __restrict__ gb, long long total) {
    long long idx = (long long)blockIdx.x * blockDim.x + threadIdx.x;
    if (idx >= total) return;
    long long g = idx >> 7;
    float v = sums[idx] / fmaxf(cnt[g], 1.0f);
    gb[idx] = f2bf(v);
}

__global__ void k_fc2(const float* __restrict__ fc1, const float* __restrict__ fcW2,
                      const float* __restrict__ fcb2, float* __restrict__ out, int G) {
    int g = blockIdx.x * blockDim.x + threadIdx.x;
    if (g >= G) return;
    float s = fcb2[0];
#pragma unroll 8
    for (int k = 0; k < HID; ++k)
        s += fc1[(long long)g * HID + k] * fcW2[k];
    out[g] = s;
}

// ---------------- host launch ----------------
extern "C" void kernel_launch(void* const* d_in, const int* in_sizes, int n_in,
                              void* d_out, int out_size, void* d_ws, size_t ws_size,
                              hipStream_t stream) {
    (void)in_sizes; (void)n_in; (void)out_size; (void)ws_size;

    const float* x     = (const float*)d_in[0];
    const int*   ei    = (const int*)d_in[1];
    const int*   src   = ei;
    const int*   dst   = ei + N_EDGES;
    const int*   batch = (const int*)d_in[2];
    const float* W1    = (const float*)d_in[3];
    const float* b1    = (const float*)d_in[4];
    const float* W2    = (const float*)d_in[5];
    const float* b2    = (const float*)d_in[6];
    const float* fcW1  = (const float*)d_in[7];
    const float* fcb1  = (const float*)d_in[8];
    const float* fcW2  = (const float*)d_in[9];
    const float* fcb2  = (const float*)d_in[10];
    float*       out   = (float*)d_out;

    // bump allocator over workspace
    char* ws = (char*)d_ws;
    size_t off = 0;
    auto alloc = [&](size_t bytes) -> void* {
        off = (off + 255) & ~(size_t)255;
        void* p = ws + off;
        off += bytes;
        return p;
    };

    float*          deg  = (float*)alloc((size_t)N_NODES * 4);                  // -> dinv in place
    unsigned short* Xb   = (unsigned short*)alloc((size_t)N_NODES * IN_DIM * 2);
    unsigned*       W1p  = (unsigned*)alloc((size_t)8 * 2 * 32 * 8 * 4);
    unsigned*       W2p  = (unsigned*)alloc((size_t)8 * 4 * 32 * 8 * 4);
    unsigned*       F1p  = (unsigned*)alloc((size_t)8 * 4 * 32 * 8 * 4);
    float*          T    = (float*)alloc((size_t)N_NODES * HID * 4);
    float*          H    = (float*)alloc((size_t)N_NODES * HID * 4);
    unsigned short* Hb   = (unsigned short*)alloc((size_t)N_NODES * HID * 2);
    float*          sums = (float*)alloc((size_t)N_GRAPHS * HID * 4);
    float*          cnt  = (float*)alloc((size_t)N_GRAPHS * 4);
    unsigned short* gb   = (unsigned short*)alloc((size_t)N_GRAPHS * HID * 2);
    float*          fc1  = (float*)alloc((size_t)N_GRAPHS * HID * 4);

    const long long NH   = (long long)N_NODES * HID;     // 12.8M
    const long long GH   = (long long)N_GRAPHS * HID;    // 262144
    const int Mt_nodes   = N_NODES / 16;                 // 6250
    const int Mt_graphs  = N_GRAPHS / 16;                // 128

    // degrees & symmetric normalization
    k_fill<<<(N_NODES + 255) / 256, 256, 0, stream>>>(deg, 1.0f, N_NODES);
    k_count_deg<<<(N_EDGES + 255) / 256, 256, 0, stream>>>(dst, deg, N_EDGES);
    k_rsqrt_inplace<<<(N_NODES + 255) / 256, 256, 0, stream>>>(deg, N_NODES);

    // bf16 casts / weight packs
    k_cast_bf16<<<(int)(((long long)N_NODES * IN_DIM + 255) / 256), 256, 0, stream>>>(
        x, Xb, (long long)N_NODES * IN_DIM);
    k_pack_w<<<16, 256, 0, stream>>>(W1, W1p, 2);
    k_pack_w<<<32, 256, 0, stream>>>(W2, W2p, 4);
    k_pack_w<<<32, 256, 0, stream>>>(fcW1, F1p, 4);

    // layer 1: T = X@W1 ; H = norm-aggregate(T) ; H = relu(H+b1), Hb = bf16(H)
    k_gemm_bf16<<<(Mt_nodes + 7) / 8, 256, 0, stream>>>(Xb, W1p, T, Mt_nodes, 2);
    k_selfloop<<<(int)((NH + 255) / 256), 256, 0, stream>>>(T, deg, H, NH);
    k_edge_agg<<<(N_EDGES + 7) / 8, 256, 0, stream>>>(src, dst, deg, T, H, N_EDGES);
    k_bias_relu_cast<<<(int)((NH + 255) / 256), 256, 0, stream>>>(H, b1, Hb, NH);

    // layer 2
    k_gemm_bf16<<<(Mt_nodes + 7) / 8, 256, 0, stream>>>(Hb, W2p, T, Mt_nodes, 4);
    k_selfloop<<<(int)((NH + 255) / 256), 256, 0, stream>>>(T, deg, H, NH);
    k_edge_agg<<<(N_EDGES + 7) / 8, 256, 0, stream>>>(src, dst, deg, T, H, N_EDGES);
    k_bias_relu_cast<<<(int)((NH + 255) / 256), 256, 0, stream>>>(H, b2, nullptr, NH);

    // global mean pool
    k_fill<<<(int)((GH + 255) / 256), 256, 0, stream>>>(sums, 0.0f, GH);
    k_fill<<<(N_GRAPHS + 255) / 256, 256, 0, stream>>>(cnt, 0.0f, N_GRAPHS);
    k_pool<<<(N_NODES + 7) / 8, 256, 0, stream>>>(batch, H, sums, cnt, N_NODES);
    k_mean_cast<<<(int)((GH + 255) / 256), 256, 0, stream>>>(sums, cnt, gb, GH);

    // FC head: fc1 = relu(g@fcW1 + fcb1); out = fc1@fcW2 + fcb2
    k_gemm_bf16<<<(Mt_graphs + 7) / 8, 256, 0, stream>>>(gb, F1p, fc1, Mt_graphs, 4);
    k_bias_relu_cast<<<(int)((GH + 255) / 256), 256, 0, stream>>>(fc1, fcb1, nullptr, GH);
    k_fc2<<<(N_GRAPHS + 255) / 256, 256, 0, stream>>>(fc1, fcW2, fcb2, out, N_GRAPHS);
}
